// VAERNN_71786083385628
// MI455X (gfx1250) — matrix operand order repaired and checked by reading, not compiled, not run on hardware
//
#include <hip/hip_runtime.h>
#include <hip/hip_bf16.h>

// VAE-RNN for MI455X (gfx1250, wave32, WMMA).
//   L=2048, B=512, K=64, IN=OBS=ST=32.
// Phase 1 (embed_kernel x2): parallel over L*B, f16 WMMA, writes xe/ue (f16) to d_ws.
// Phase 2 (scan_kernel): 32 blocks, each owns 16 batch rows (WMMA M=16) and walks
//   the 2048-step recurrence with h in LDS. Weight fragments live in VGPRs across
//   the whole loop. All matmuls are V_WMMA_F32_16X16X32_F16 (f16 in, f32 acc).
// Workspace requirement: 2 * L*B*64 * 2 bytes = 256 MiB.

typedef __attribute__((ext_vector_type(16))) _Float16 v16h;
typedef __attribute__((ext_vector_type(8)))  float    v8f;

#define LSEQ  2048
#define BATCH 512

#define LOG2E      1.4426950408889634f
#define TWO_LOG2E  2.8853900817779268f

union FragU { uint4 u[2]; v16h v; };

__device__ __forceinline__ v8f wmma_f16(v16h a, v16h b, v8f c) {
  // D = A(16x32 f16) * B(32x16 f16) + C(16x16 f32)
  return __builtin_amdgcn_wmma_f32_16x16x32_f16(
      /*neg_a=*/false, a, /*neg_b=*/false, b,
      /*c_mod=*/(short)0, c, /*reuse_a=*/false, /*reuse_b=*/false);
}

// ---- fast branch-free activations (v_fma + v_exp_f32 + v_rcp_f32) ----
__device__ __forceinline__ float rcp_fast(float x) {
  return __builtin_amdgcn_rcpf(x);
}
__device__ __forceinline__ float exp2_fast(float x) {
  return __builtin_amdgcn_exp2f(x);       // raw v_exp_f32
}
__device__ __forceinline__ float sigmoid_fast(float x) {
  return rcp_fast(1.0f + exp2_fast(-LOG2E * x));
}
__device__ __forceinline__ float tanh_fast(float x) {
  // tanh(x) = 1 - 2/(exp(2x)+1); saturates correctly for |x| large.
  return 1.0f - 2.0f * rcp_fast(1.0f + exp2_fast(TWO_LOG2E * x));
}
// tanh(c + b) with pre-scaled bias bs = b * 2*log2e  (single fma before exp)
__device__ __forceinline__ float tanh_biased(float c, float bs) {
  return 1.0f - 2.0f * rcp_fast(1.0f + exp2_fast(__builtin_fmaf(c, TWO_LOG2E, bs)));
}
// sigmoid(c + b) with pre-scaled bias bs = -b * log2e
__device__ __forceinline__ float sigmoid_biased(float c, float bs) {
  return rcp_fast(1.0f + exp2_fast(__builtin_fmaf(c, -LOG2E, bs)));
}

// ---- fragment builders (layouts per CDNA5 ISA §7.12.2) ----

// A (16x32 f16) from LDS row-major [16][ld] starting at column k0.
// lane l: m=l&15; halves 0..7 -> k = k0 + 8*(l>=16) + i ; halves 8..15 -> +16.
__device__ __forceinline__ v16h a_frag_lds(const _Float16* base, int ld, int k0) {
  int lane = threadIdx.x & 31;
  const _Float16* p = base + (lane & 15) * ld + k0 + ((lane >> 4) * 8);
  FragU r;
  r.u[0] = *(const uint4*)(p);        // halves k..k+7
  r.u[1] = *(const uint4*)(p + 16);   // halves k+16..k+23
  return r.v;
}

// A (16x32 f16) gathered from global f32 matrix X [rows][ld], rows row0..row0+15.
__device__ __forceinline__ v16h a_frag_g(const float* X, int ld, long row0, int k0) {
  int lane = threadIdx.x & 31;
  const float* p = X + (size_t)(row0 + (lane & 15)) * ld + k0 + ((lane >> 4) * 8);
  v16h a;
#pragma unroll
  for (int i = 0; i < 8; ++i) a[i] = (_Float16)p[i];
#pragma unroll
  for (int i = 0; i < 8; ++i) a[8 + i] = (_Float16)p[16 + i];
  return a;
}

// B (32x16 f16) = W.T tile from global f32 W [n][ldk] (i.e., B[k][n] = W[n][k]).
// lane l: n = n0 + (l&15); halves 0..7 -> k = k0 + 8*(l>=16) + i ; halves 8..15 -> +16.
__device__ __forceinline__ v16h b_frag_g(const float* W, int ldk, int n0, int k0) {
  int lane = threadIdx.x & 31;
  const float* p = W + (size_t)(n0 + (lane & 15)) * ldk + k0 + ((lane >> 4) * 8);
  v16h b;
#pragma unroll
  for (int i = 0; i < 8; ++i) b[i] = (_Float16)p[i];
#pragma unroll
  for (int i = 0; i < 8; ++i) b[8 + i] = (_Float16)p[16 + i];
  return b;
}

// ---------------------------------------------------------------------------
// Kernel A: two-layer tanh MLP embedding.  X:[L*B,32] f32 -> OUT:[L*B,64] f16
// ---------------------------------------------------------------------------
__global__ __launch_bounds__(128) void embed_kernel(
    const float* __restrict__ X,
    const float* __restrict__ w1, const float* __restrict__ b1,   // (64,32),(64)
    const float* __restrict__ w2, const float* __restrict__ b2,   // (64,64),(64)
    _Float16* __restrict__ OUT)
{
  __shared__ __align__(16) _Float16 stage[4][16 * 64];  // per-wave hidden tile
  const int wv   = threadIdx.x >> 5;
  const int lane = threadIdx.x & 31;
  const int ncol = lane & 15;
  const int mb   = (lane >> 4) * 8;
  _Float16* myT  = stage[wv];

  // loop-invariant weight fragments (held in VGPRs)
  v16h B1[4], B2[4][2];
#pragma unroll
  for (int j = 0; j < 4; ++j) {
    B1[j]    = b_frag_g(w1, 32, j * 16, 0);
    B2[j][0] = b_frag_g(w2, 64, j * 16, 0);
    B2[j][1] = b_frag_g(w2, 64, j * 16, 32);
  }
  float b1s[4], b2s[4];   // biases pre-scaled by 2*log2e for tanh_biased
#pragma unroll
  for (int j = 0; j < 4; ++j) {
    b1s[j] = b1[j * 16 + ncol] * TWO_LOG2E;
    b2s[j] = b2[j * 16 + ncol] * TWO_LOG2E;
  }

  const long nTiles = (long)LSEQ * BATCH / 16;
  const long gw     = (long)blockIdx.x * 4 + wv;
  const long stride = (long)gridDim.x * 4;

  for (long t = gw; t < nTiles; t += stride) {
    const long row0 = t * 16;
    // prefetch next tile's input rows (2 KB contiguous)
    if (t + stride < nTiles) {
      const float* np = X + (size_t)(t + stride) * 16 * 32;
      __builtin_prefetch(np + lane * 16, 0, 0);
    }
    // layer 1: (16x32) @ (32x64), K=32 -> one WMMA per 16-wide N tile
    v16h a1 = a_frag_g(X, 32, row0, 0);
#pragma unroll
    for (int j = 0; j < 4; ++j) {
      v8f c = {};
      c = wmma_f16(a1, B1[j], c);
#pragma unroll
      for (int r = 0; r < 8; ++r)
        myT[(mb + r) * 64 + j * 16 + ncol] = (_Float16)tanh_biased(c[r], b1s[j]);
    }
    // layer 2: (16x64) @ (64x64), K=64 -> two chained WMMAs per N tile
    v16h a20 = a_frag_lds(myT, 64, 0);
    v16h a21 = a_frag_lds(myT, 64, 32);
#pragma unroll
    for (int j = 0; j < 4; ++j) {
      v8f c = {};
      c = wmma_f16(a20, B2[j][0], c);
      c = wmma_f16(a21, B2[j][1], c);
#pragma unroll
      for (int r = 0; r < 8; ++r)
        OUT[(size_t)(row0 + mb + r) * 64 + j * 16 + ncol] =
            (_Float16)tanh_biased(c[r], b2s[j]);
    }
  }
}

// ---------------------------------------------------------------------------
// Kernel B: sequential scan. One block = 16 batch rows, 8 waves, 2048 steps.
// ---------------------------------------------------------------------------
__global__ __launch_bounds__(256) void scan_kernel(
    const _Float16* __restrict__ xeS, const _Float16* __restrict__ ueS,
    const float* __restrict__ eps,
    const float* __restrict__ wf1, const float* __restrict__ bf1v,
    const float* __restrict__ wf2, const float* __restrict__ bf2v,
    const float* __restrict__ wmu, const float* __restrict__ bmu,
    const float* __restrict__ wls, const float* __restrict__ bls,
    const float* __restrict__ wih, const float* __restrict__ whh,
    const float* __restrict__ bih, const float* __restrict__ bhh,
    float* __restrict__ out_mu, float* __restrict__ out_ls,
    float* __restrict__ out_z,  float* __restrict__ out_h,
    float* __restrict__ out_hn)
{
  // LDS (~44.5 KB static)
  __shared__ __align__(16) _Float16 catL[16 * 128];  // [xe | h] f16 (A for DBlock)
  __shared__ __align__(16) _Float16 ueL[16 * 64];    // ue f16
  __shared__ __align__(16) _Float16 tL[16 * 128];    // gated t f16
  __shared__ __align__(16) float    hnL[16 * 64];    // h f32 (master copy)
  __shared__ __align__(16) float    gigL[16 * 384];  // [gi(192) | gh(192)] f32
  __shared__ __align__(16) float    mlsL[16 * 64];   // [mu(32) | ls(32)] f32
  __shared__ __align__(16) float    bihL[192];
  __shared__ __align__(16) float    bhhL[192];

  const int tid  = threadIdx.x;
  const int wv   = tid >> 5;          // wave 0..7
  const int lane = tid & 31;
  const int ncol = lane & 15;
  const int mb   = (lane >> 4) * 8;
  const int b0   = blockIdx.x * 16;   // batch rows b0..b0+15

  // ---- loop-invariant weight fragments in VGPRs ----
  // Stage 1: every wave owns N-tile `wv` of both f1 and f2 (K = 2K = 128).
  v16h Bf1[4], Bf2[4];
#pragma unroll
  for (int ch = 0; ch < 4; ++ch) {
    Bf1[ch] = b_frag_g(wf1, 128, wv * 16, ch * 32);
    Bf2[ch] = b_frag_g(wf2, 128, wv * 16, ch * 32);
  }
  // Stage 2a (waves 0-3): combined [mu|ls] N-tiles, K=128.
  v16h Bml[4] = {};
  float bMLr = 0.0f;
  if (wv < 4) {
    const float* W = (wv < 2) ? wmu : wls;
    const int n0   = (wv & 1) * 16;
#pragma unroll
    for (int ch = 0; ch < 4; ++ch) Bml[ch] = b_frag_g(W, 128, n0, ch * 32);
    bMLr = (wv < 2) ? bmu[wv * 16 + ncol] : bls[(wv - 2) * 16 + ncol];
  }
  // Stage 2b (waves 4-7): 3 N-tiles each of wih and whh, K=64.
  v16h Bih[3][2] = {}, Bhh[3][2] = {};
  if (wv >= 4) {
    const int w2 = wv - 4;
#pragma unroll
    for (int j = 0; j < 3; ++j) {
      const int n0 = (w2 * 3 + j) * 16;
      Bih[j][0] = b_frag_g(wih, 64, n0, 0);
      Bih[j][1] = b_frag_g(wih, 64, n0, 32);
      Bhh[j][0] = b_frag_g(whh, 64, n0, 0);
      Bhh[j][1] = b_frag_g(whh, 64, n0, 32);
    }
  }
  const float bF1s = bf1v[wv * 16 + ncol] * TWO_LOG2E;  // pre-scaled for tanh
  const float bF2s = bf2v[wv * 16 + ncol] * -LOG2E;     // pre-scaled for sigmoid

  // ---- init h = 0, cache GRU biases ----
  for (int i = tid; i < 16 * 64; i += 256) hnL[i] = 0.0f;
  for (int i = tid; i < 16 * 64; i += 256) {
    int m = i >> 6, c = i & 63;
    catL[m * 128 + 64 + c] = (_Float16)0.0f;
  }
  for (int i = tid; i < 192; i += 256) { bihL[i] = bih[i]; bhhL[i] = bhh[i]; }
  __syncthreads();

  for (int l = 0; l < LSEQ; ++l) {
    // ---- load this step's xe / ue tiles (16 rows x 64 f16 each, b128 each) ----
    {
      const size_t rbase = ((size_t)l * BATCH + b0) * 64;
      if (tid < 128) {                       // waves 0-3: xe -> catL[:, 0:64]
        const int i = tid * 8;
        const int m = i >> 6, k = i & 63;
        *(uint4*)&catL[m * 128 + k] = *(const uint4*)&xeS[rbase + i];
      } else {                               // waves 4-7: ue -> ueL
        const int i = (tid - 128) * 8;
        *(uint4*)&ueL[i] = *(const uint4*)&ueS[rbase + i];
      }
      // prefetch next step's tiles while this step computes
      if (l + 1 < LSEQ) {
        const size_t nbase = rbase + (size_t)BATCH * 64;
        const int i = tid * 4;
        __builtin_prefetch(&xeS[nbase + i], 0, 0);
        __builtin_prefetch(&ueS[nbase + i], 0, 0);
        if (tid < 64)
          __builtin_prefetch(&eps[((size_t)(l + 1) * BATCH + b0) * 32 + tid * 8], 0, 0);
      }
    }
    __syncthreads();

    // ---- Stage 1: f1/f2 = [xe|h] @ {wf1,wf2}.T ; t = tanh(f1)*sigmoid(f2) ----
    {
      v16h ac0 = a_frag_lds(catL, 128, 0);
      v16h ac1 = a_frag_lds(catL, 128, 32);
      v16h ac2 = a_frag_lds(catL, 128, 64);
      v16h ac3 = a_frag_lds(catL, 128, 96);
      v8f c1 = {}, c2 = {};
      c1 = wmma_f16(ac0, Bf1[0], c1); c1 = wmma_f16(ac1, Bf1[1], c1);
      c1 = wmma_f16(ac2, Bf1[2], c1); c1 = wmma_f16(ac3, Bf1[3], c1);
      c2 = wmma_f16(ac0, Bf2[0], c2); c2 = wmma_f16(ac1, Bf2[1], c2);
      c2 = wmma_f16(ac2, Bf2[2], c2); c2 = wmma_f16(ac3, Bf2[3], c2);
#pragma unroll
      for (int r = 0; r < 8; ++r) {
        float tv = tanh_biased(c1[r], bF1s) * sigmoid_biased(c2[r], bF2s);
        tL[(mb + r) * 128 + wv * 16 + ncol] = (_Float16)tv;
      }
    }
    __syncthreads();

    // ---- Stage 2: waves 0-3 -> mu/ls ; waves 4-7 -> gi/gh ----
    if (wv < 4) {
      v16h at0 = a_frag_lds(tL, 128, 0);
      v16h at1 = a_frag_lds(tL, 128, 32);
      v16h at2 = a_frag_lds(tL, 128, 64);
      v16h at3 = a_frag_lds(tL, 128, 96);
      v8f cm = {};
      cm = wmma_f16(at0, Bml[0], cm); cm = wmma_f16(at1, Bml[1], cm);
      cm = wmma_f16(at2, Bml[2], cm); cm = wmma_f16(at3, Bml[3], cm);
#pragma unroll
      for (int r = 0; r < 8; ++r)
        mlsL[(mb + r) * 64 + wv * 16 + ncol] = cm[r] + bMLr;
    } else {
      v16h au0 = a_frag_lds(ueL, 64, 0);
      v16h au1 = a_frag_lds(ueL, 64, 32);
      v16h ah0 = a_frag_lds(catL + 64, 128, 0);   // h columns of cat
      v16h ah1 = a_frag_lds(catL + 64, 128, 32);
      const int w2 = wv - 4;
#pragma unroll
      for (int j = 0; j < 3; ++j) {
        const int n0 = (w2 * 3 + j) * 16;
        v8f cg = {};
        cg = wmma_f16(au0, Bih[j][0], cg);
        cg = wmma_f16(au1, Bih[j][1], cg);
#pragma unroll
        for (int r = 0; r < 8; ++r) gigL[(mb + r) * 384 + n0 + ncol] = cg[r];
        v8f chv = {};
        chv = wmma_f16(ah0, Bhh[j][0], chv);
        chv = wmma_f16(ah1, Bhh[j][1], chv);
#pragma unroll
        for (int r = 0; r < 8; ++r) gigL[(mb + r) * 384 + 192 + n0 + ncol] = chv[r];
      }
    }
    __syncthreads();

    // ---- element-wise: reparameterized sample (float2 lanes) ----
    {
      const int i = tid * 2;               // 512 elems / 256 threads
      const int m = i >> 5, c = i & 31;
      const float2 mu2 = *(const float2*)&mlsL[m * 64 + c];
      const float2 ls2 = *(const float2*)&mlsL[m * 64 + 32 + c];
      const size_t g = ((size_t)l * BATCH + b0 + m) * 32 + c;
      const float2 ep2 = *(const float2*)&eps[g];
      float2 z2;
      z2.x = mu2.x + __expf(ls2.x) * ep2.x;
      z2.y = mu2.y + __expf(ls2.y) * ep2.y;
      *(float2*)&out_mu[g] = mu2;
      *(float2*)&out_ls[g] = ls2;
      *(float2*)&out_z[g]  = z2;
    }
    // ---- element-wise: GRU update (float4 lanes) ----
    {
      const int i = tid * 4;               // 1024 elems / 256 threads
      const int m = i >> 6, c = i & 63;
      const float4 h4  = *(const float4*)&hnL[i];
      const float4 vir = *(const float4*)&gigL[m * 384 +        c];
      const float4 viz = *(const float4*)&gigL[m * 384 +  64 + c];
      const float4 vig = *(const float4*)&gigL[m * 384 + 128 + c];
      const float4 vhr = *(const float4*)&gigL[m * 384 + 192 + c];
      const float4 vhz = *(const float4*)&gigL[m * 384 + 256 + c];
      const float4 vhg = *(const float4*)&gigL[m * 384 + 320 + c];
      const float4 bir = *(const float4*)&bihL[c];
      const float4 biz = *(const float4*)&bihL[64 + c];
      const float4 big = *(const float4*)&bihL[128 + c];
      const float4 bhr = *(const float4*)&bhhL[c];
      const float4 bhz = *(const float4*)&bhhL[64 + c];
      const float4 bhg = *(const float4*)&bhhL[128 + c];
      float4 hnew4;
      _Float16 hh[4];
      const float* hp   = (const float*)&h4;
      const float* pir  = (const float*)&vir;
      const float* piz  = (const float*)&viz;
      const float* pig  = (const float*)&vig;
      const float* phr  = (const float*)&vhr;
      const float* phz  = (const float*)&vhz;
      const float* phg  = (const float*)&vhg;
      const float* pbir = (const float*)&bir;
      const float* pbiz = (const float*)&biz;
      const float* pbig = (const float*)&big;
      const float* pbhr = (const float*)&bhr;
      const float* pbhz = (const float*)&bhz;
      const float* pbhg = (const float*)&bhg;
      float* pnew = (float*)&hnew4;
#pragma unroll
      for (int q = 0; q < 4; ++q) {
        const float h  = hp[q];
        const float rg = sigmoid_fast((pir[q] + pbir[q]) + (phr[q] + pbhr[q]));
        const float zg = sigmoid_fast((piz[q] + pbiz[q]) + (phz[q] + pbhz[q]));
        const float hg = phg[q] + pbhg[q];
        const float ng = tanh_fast((pig[q] + pbig[q]) + rg * hg);
        const float hnew = (1.0f - zg) * ng + zg * h;
        pnew[q] = hnew;
        hh[q]   = (_Float16)hnew;
      }
      *(float4*)&out_h[((size_t)l * BATCH + b0 + m) * 64 + c] = h4;  // h BEFORE update
      *(float4*)&hnL[i] = hnew4;
      *(uint2*)&catL[m * 128 + 64 + c] = *(uint2*)hh;
    }
    __syncthreads();
  }

  // final hidden state (float4 lanes)
  {
    const int i = tid * 4;
    const int m = i >> 6, c = i & 63;
    *(float4*)&out_hn[(size_t)(b0 + m) * 64 + c] = *(const float4*)&hnL[i];
  }
}

// ---------------------------------------------------------------------------
extern "C" void kernel_launch(void* const* d_in, const int* in_sizes, int n_in,
                              void* d_out, int out_size, void* d_ws, size_t ws_size,
                              hipStream_t stream) {
  (void)in_sizes; (void)n_in; (void)out_size; (void)ws_size;

  const float* ext = (const float*)d_in[0];
  const float* obs = (const float*)d_in[1];
  const float* eps = (const float*)d_in[2];
  const float* pu_w1 = (const float*)d_in[3];
  const float* pu_b1 = (const float*)d_in[4];
  const float* pu_w2 = (const float*)d_in[5];
  const float* pu_b2 = (const float*)d_in[6];
  const float* px_w1 = (const float*)d_in[7];
  const float* px_b1 = (const float*)d_in[8];
  const float* px_w2 = (const float*)d_in[9];
  const float* px_b2 = (const float*)d_in[10];
  const float* wf1 = (const float*)d_in[11];
  const float* bf1 = (const float*)d_in[12];
  const float* wf2 = (const float*)d_in[13];
  const float* bf2 = (const float*)d_in[14];
  const float* wmu = (const float*)d_in[15];
  const float* bmu = (const float*)d_in[16];
  const float* wls = (const float*)d_in[17];
  const float* bls = (const float*)d_in[18];
  const float* wih = (const float*)d_in[19];
  const float* whh = (const float*)d_in[20];
  const float* bih = (const float*)d_in[21];
  const float* bhh = (const float*)d_in[22];

  const size_t LB = (size_t)LSEQ * BATCH;
  // workspace: xe (f16) then ue (f16); needs 2*LB*64*2 = 256 MiB
  _Float16* xeS = (_Float16*)d_ws;
  _Float16* ueS = xeS + LB * 64;

  float* out_mu = (float*)d_out;
  float* out_ls = out_mu + LB * 32;
  float* out_z  = out_ls + LB * 32;
  float* out_h  = out_z  + LB * 32;
  float* out_hn = out_h  + LB * 64;

  embed_kernel<<<2048, 128, 0, stream>>>(obs, px_w1, px_b1, px_w2, px_b2, xeS);
  embed_kernel<<<2048, 128, 0, stream>>>(ext, pu_w1, pu_b1, pu_w2, pu_b2, ueS);
  scan_kernel<<<BATCH / 16, 256, 0, stream>>>(
      xeS, ueS, eps,
      wf1, bf1, wf2, bf2, wmu, bmu, wls, bls,
      wih, whh, bih, bhh,
      out_mu, out_ls, out_z, out_h, out_hn);
}
